// ConvGenerator_52999896432809
// MI455X (gfx1250) — matrix operand order repaired and checked by reading, not compiled, not run on hardware
//
#include <hip/hip_runtime.h>
#include <hip/hip_bf16.h>

typedef __attribute__((ext_vector_type(16))) _Float16 v16h;
typedef __attribute__((ext_vector_type(8)))  float    v8f;

#define GK    25
#define NN    8192
#define EN    131072
#define N2C   4096
#define E2C   65536
#define BG    64
#define EPSC  1e-15f

// ---------------------------------------------------------------------------
// A-fragment loader (CDNA5 16-bit A layout, ISA 7.12.2), branch-free:
// pair (ka, ka+1) lives in 16-aligned K block b = (j<4 ? 0 : 1), so validity
// (k0 + b*16 + 16 <= KMAX) is half-independent and folds at compile time.
// Loads are 4x float4 per k-step.
// ---------------------------------------------------------------------------
template <int KMAX>
__device__ __forceinline__ v16h load_a_frag(const float* __restrict__ rowptr,
                                            int k0, int half) {
  v16h a;
  if (k0 + 16 <= KMAX) {                       // K block [k0, k0+16)
    const float4* p = (const float4*)(rowptr + k0 + 8 * half);
    float4 q0 = p[0], q1 = p[1];
    a[0] = (_Float16)q0.x; a[1] = (_Float16)q0.y;
    a[2] = (_Float16)q0.z; a[3] = (_Float16)q0.w;
    a[4] = (_Float16)q1.x; a[5] = (_Float16)q1.y;
    a[6] = (_Float16)q1.z; a[7] = (_Float16)q1.w;
  } else {
#pragma unroll
    for (int t = 0; t < 8; ++t) a[t] = (_Float16)0.0f;
  }
  if (k0 + 32 <= KMAX) {                       // K block [k0+16, k0+32)
    const float4* p = (const float4*)(rowptr + k0 + 16 + 8 * half);
    float4 q0 = p[0], q1 = p[1];
    a[8]  = (_Float16)q0.x; a[9]  = (_Float16)q0.y;
    a[10] = (_Float16)q0.z; a[11] = (_Float16)q0.w;
    a[12] = (_Float16)q1.x; a[13] = (_Float16)q1.y;
    a[14] = (_Float16)q1.z; a[15] = (_Float16)q1.w;
  } else {
#pragma unroll
    for (int t = 8; t < 16; ++t) a[t] = (_Float16)0.0f;
  }
  return a;
}

// ---------------------------------------------------------------------------
// Pre-pack a row-major f32 weight matrix B[IN_, NC] into f16 B-fragments:
// flat f16 index i = (((nt*NKS)+ks)*32 + lane)*16 + e, with
// e  ->  K = ks*32 + (e&7) + 8*(lane>>4) + 16*(e>>3), col = nt*16 + (lane&15).
// Zero-pads K >= IN_. Consumer loads one contiguous v16h per lane.
// ---------------------------------------------------------------------------
__global__ void pack_b_frags(const float* __restrict__ B, _Float16* __restrict__ Bp,
                             int IN_, int NC, int NKS, int total) {
  int i = blockIdx.x * blockDim.x + threadIdx.x;
  if (i >= total) return;
  int e    = i & 15;
  int lane = (i >> 4) & 31;
  int t    = i >> 9;
  int ks   = t % NKS, nt = t / NKS;
  int half = lane >> 4, n = lane & 15;
  int j = e >> 1, lo = e & 1;
  int kr = ks * 32 + (e & 7) + 8 * half + 16 * (e >> 3);
  float v = (kr < IN_) ? B[(long)kr * NC + nt * 16 + n] : 0.0f;
  Bp[i] = (_Float16)v;
}

// ---------------------------------------------------------------------------
// Gaussian mixture kernel weights
// ---------------------------------------------------------------------------
__global__ void gauss_kernel(const float* __restrict__ ea, const float* __restrict__ mu,
                             const float* __restrict__ sigma, float* __restrict__ gauss,
                             int total) {
  int i = blockIdx.x * blockDim.x + threadIdx.x;
  if (i >= total) return;
  int e = i / GK, k = i % GK;
  float d0 = ea[e * 2 + 0] - mu[k * 2 + 0];
  float d1 = ea[e * 2 + 1] - mu[k * 2 + 1];
  float s0 = sigma[k * 2 + 0], s1 = sigma[k * 2 + 1];
  float q = d0 * d0 / (EPSC + s0 * s0) + d1 * d1 / (EPSC + s1 * s1);
  gauss[i] = __expf(-0.5f * q);
}

// ---------------------------------------------------------------------------
// Fused GMMConv edge pass: wave <-> 16 edges; A = gathered x[src] rows,
// B = pre-packed fragments of g; contract y-tiles with gauss in registers;
// atomic scatter to agg[dst].
// ---------------------------------------------------------------------------
template <int IN, int OUT>
__global__ void __launch_bounds__(256)
gmm_edge_wmma(const float* __restrict__ X, const int* __restrict__ ei,
              const float* __restrict__ gauss, const _Float16* __restrict__ Gp,
              float* __restrict__ agg, float* __restrict__ cnt, int nedges) {
  constexpr int OB  = OUT / 16;
  constexpr int NKS = (IN + 31) / 32;
  __shared__ float gs[8][16 * GK];
  __shared__ int   dsts[8][16];

  const int lane = threadIdx.x & 31;
  const int wave = threadIdx.x >> 5;
  const int half = lane >> 4;
  const int mr   = lane & 15;
  const int myedge = (blockIdx.x * 8 + wave) * 16 + mr;

  const int src = ei[myedge];
  if (half == 0) {
    dsts[wave][mr] = ei[nedges + myedge];
    for (int k = 0; k < GK; ++k) gs[wave][mr * GK + k] = gauss[myedge * GK + k];
  }
  __syncthreads();

  v16h afrag[NKS];
  const float* xrow = X + (long)src * IN;
#pragma unroll
  for (int ks = 0; ks < NKS; ++ks) afrag[ks] = load_a_frag<IN>(xrow, ks * 32, half);

  const v16h* Gf = (const v16h*)Gp;
  float acc[OB][8];
#pragma unroll
  for (int ob = 0; ob < OB; ++ob)
#pragma unroll
    for (int r = 0; r < 8; ++r) acc[ob][r] = 0.0f;

  for (int kk = 0; kk < GK; ++kk) {
    float gv[8];
#pragma unroll
    for (int r = 0; r < 8; ++r) gv[r] = gs[wave][(r + 8 * half) * GK + kk];
#pragma unroll
    for (int ob = 0; ob < OB; ++ob) {
      v8f y = {};
#pragma unroll
      for (int ks = 0; ks < NKS; ++ks) {
        v16h b = Gf[(size_t)(((kk * OB + ob) * NKS) + ks) * 32 + lane];
        y = __builtin_amdgcn_wmma_f32_16x16x32_f16(false, afrag[ks], false, b,
                                                   (short)0, y, false, false);
      }
#pragma unroll
      for (int r = 0; r < 8; ++r) acc[ob][r] += gv[r] * y[r];
    }
  }

#pragma unroll
  for (int ob = 0; ob < OB; ++ob)
#pragma unroll
    for (int r = 0; r < 8; ++r) {
      const int m = r + 8 * half;
      atomicAdd(&agg[(long)dsts[wave][m] * OUT + ob * 16 + mr], acc[ob][r]);
    }
  if (half == 0) atomicAdd(&cnt[dsts[wave][mr]], 1.0f);
}

// ---------------------------------------------------------------------------
// Root-weight GEMM via WMMA with pre-packed W fragments
// ---------------------------------------------------------------------------
template <int IN, int OUT>
__global__ void __launch_bounds__(32)
gemm_root_wmma(const float* __restrict__ X, const _Float16* __restrict__ Wp,
               float* __restrict__ R) {
  constexpr int NKS = (IN + 31) / 32;
  const int lane = threadIdx.x & 31;
  const int half = lane >> 4;
  const int mr   = lane & 15;
  const int nt   = blockIdx.x, mt = blockIdx.y;
  const int ncol = nt * 16 + mr;
  const float* xrow = X + (long)(mt * 16 + mr) * IN;
  const v16h* Wf = (const v16h*)Wp;
  v8f c = {};
#pragma unroll
  for (int ks = 0; ks < NKS; ++ks) {
    v16h a = load_a_frag<IN>(xrow, ks * 32, half);
    v16h b = Wf[(size_t)(nt * NKS + ks) * 32 + lane];
    c = __builtin_amdgcn_wmma_f32_16x16x32_f16(false, a, false, b, (short)0, c,
                                               false, false);
  }
#pragma unroll
  for (int r = 0; r < 8; ++r) {
    const int m = mt * 16 + r + 8 * half;
    R[(long)m * OUT + ncol] = c[r];
  }
}

// h = elu(agg/max(cnt,1) + root_term + bias)
template <int OUT>
__global__ void finalize_elu(const float* __restrict__ agg, const float* __restrict__ cnt,
                             const float* __restrict__ rterm, const float* __restrict__ bias,
                             float* __restrict__ h, int total) {
  int i = blockIdx.x * blockDim.x + threadIdx.x;
  if (i >= total) return;
  int n = i / OUT, o = i % OUT;
  float c = cnt[n]; c = c < 1.0f ? 1.0f : c;
  float v = agg[i] / c + rterm[i] + bias[o];
  h[i] = v > 0.0f ? v : (__expf(v) - 1.0f);
}

// ------------------------- pooling -----------------------------------------
__device__ __forceinline__ unsigned f32_to_ordered_u32(float f) {
  int b = __float_as_int(f);
  return (b >= 0) ? ((unsigned)b | 0x80000000u) : ~(unsigned)b;
}

__global__ void pool_max_scatter(const float* __restrict__ h, const int* __restrict__ cluster,
                                 unsigned* __restrict__ umax, int total) {
  int i = blockIdx.x * blockDim.x + threadIdx.x;
  if (i >= total) return;
  int n = i / 64, o = i % 64;
  atomicMax(&umax[(long)cluster[n] * 64 + o], f32_to_ordered_u32(h[i]));
}

__global__ void pool_max_decode(const unsigned* __restrict__ umax, float* __restrict__ h2,
                                int total) {
  int i = blockIdx.x * blockDim.x + threadIdx.x;
  if (i >= total) return;
  unsigned u = umax[i];
  float v;
  if (u == 0u) v = 0.0f;                                        // empty cluster
  else if (u & 0x80000000u) v = __int_as_float((int)(u & 0x7FFFFFFFu));
  else v = __int_as_float((int)~u);
  h2[i] = v;
}

__global__ void pos_scatter(const float* __restrict__ pos, const int* __restrict__ cluster,
                            float* __restrict__ psum, float* __restrict__ pcnt, int n) {
  int i = blockIdx.x * blockDim.x + threadIdx.x;
  if (i >= n) return;
  int c = cluster[i];
  atomicAdd(&psum[c * 2 + 0], pos[i * 2 + 0]);
  atomicAdd(&psum[c * 2 + 1], pos[i * 2 + 1]);
  atomicAdd(&pcnt[c], 1.0f);
}

__global__ void pos_mean(const float* __restrict__ psum, const float* __restrict__ pcnt,
                         float* __restrict__ pos2, int n2) {
  int i = blockIdx.x * blockDim.x + threadIdx.x;
  if (i >= n2) return;
  float c = pcnt[i]; c = c < 1.0f ? 1.0f : c;
  pos2[i * 2 + 0] = psum[i * 2 + 0] / c;
  pos2[i * 2 + 1] = psum[i * 2 + 1] / c;
}

__global__ void cart_max(const float* __restrict__ pos2, const int* __restrict__ ei2,
                         unsigned* __restrict__ mx, int ne) {
  int e = blockIdx.x * blockDim.x + threadIdx.x;
  if (e >= ne) return;
  int s = ei2[e], d = ei2[ne + e];
  float c0 = pos2[d * 2 + 0] - pos2[s * 2 + 0];
  float c1 = pos2[d * 2 + 1] - pos2[s * 2 + 1];
  float a = fmaxf(fabsf(c0), fabsf(c1));
  atomicMax(mx, (unsigned)__float_as_int(a)); // non-negative floats order as ints
}

__global__ void cart_write(const float* __restrict__ pos2, const int* __restrict__ ei2,
                           const unsigned* __restrict__ mx, float* __restrict__ ea2, int ne) {
  int e = blockIdx.x * blockDim.x + threadIdx.x;
  if (e >= ne) return;
  float m = __int_as_float((int)*mx);
  float inv = 1.0f / (2.0f * m);
  int s = ei2[e], d = ei2[ne + e];
  ea2[e * 2 + 0] = (pos2[d * 2 + 0] - pos2[s * 2 + 0]) * inv + 0.5f;
  ea2[e * 2 + 1] = (pos2[d * 2 + 1] - pos2[s * 2 + 1]) * inv + 0.5f;
}

// ------------------------- head --------------------------------------------
__global__ void graph_scatter(const float* __restrict__ h3, const int* __restrict__ batch2,
                              float* __restrict__ gsum, float* __restrict__ gcnt, int total) {
  int i = blockIdx.x * blockDim.x + threadIdx.x;
  if (i >= total) return;
  int n = i / 64, o = i % 64;
  int b = batch2[n];
  atomicAdd(&gsum[b * 64 + o], h3[i]);
  if (o == 0) atomicAdd(&gcnt[b], 1.0f);
}

__global__ void __launch_bounds__(128)
head_kernel(const float* __restrict__ gsum, const float* __restrict__ gcnt,
            const float* __restrict__ w1, const float* __restrict__ b1,
            const float* __restrict__ w2, const float* __restrict__ b2,
            float* __restrict__ out) {
  __shared__ float hg[64];
  __shared__ float t[128];
  int b = blockIdx.x, j = threadIdx.x;
  if (j < 64) {
    float c = gcnt[b]; c = c < 1.0f ? 1.0f : c;
    hg[j] = gsum[b * 64 + j] / c;
  }
  __syncthreads();
  float s = b1[j];
  for (int i = 0; i < 64; ++i) s += hg[i] * w1[i * 128 + j];
  t[j] = s > 0.0f ? s : (__expf(s) - 1.0f);
  __syncthreads();
  if (j < 3) {
    float s2 = b2[j];
    for (int i = 0; i < 128; ++i) s2 += t[i] * w2[i * 3 + j];
    out[b * 3 + j] = 1.0f / (1.0f + __expf(-s2));
  }
}

// ---------------------------------------------------------------------------
extern "C" void kernel_launch(void* const* d_in, const int* in_sizes, int n_in,
                              void* d_out, int out_size, void* d_ws, size_t ws_size,
                              hipStream_t stream) {
  (void)in_sizes; (void)n_in; (void)out_size; (void)ws_size;
  const float* x      = (const float*)d_in[0];
  const float* pos    = (const float*)d_in[1];
  const int*   ei     = (const int*)d_in[2];
  const float* eattr  = (const float*)d_in[3];
  const int*   clus   = (const int*)d_in[4];
  const int*   ei2    = (const int*)d_in[5];
  const int*   batch2 = (const int*)d_in[6];
  const float* g1 = (const float*)d_in[7],  *mu1 = (const float*)d_in[8],
             *sg1 = (const float*)d_in[9],  *rt1 = (const float*)d_in[10],
             *bb1 = (const float*)d_in[11];
  const float* g2 = (const float*)d_in[12], *mu2 = (const float*)d_in[13],
             *sg2 = (const float*)d_in[14], *rt2 = (const float*)d_in[15],
             *bb2 = (const float*)d_in[16];
  const float* g3 = (const float*)d_in[17], *mu3 = (const float*)d_in[18],
             *sg3 = (const float*)d_in[19], *rt3 = (const float*)d_in[20],
             *bb3 = (const float*)d_in[21];
  const float* w1 = (const float*)d_in[22], *b1v = (const float*)d_in[23];
  const float* w2 = (const float*)d_in[24], *b2v = (const float*)d_in[25];
  float* out = (float*)d_out;

  // workspace carve-out (floats, 256B aligned)
  float* ws = (float*)d_ws;
  size_t off = 0;
  auto carve  = [&](size_t nf) { float* p = ws + off; off += (nf + 63) & ~(size_t)63; return p; };
  auto carveh = [&](size_t nh) { return (_Float16*)carve((nh + 1) / 2); };
  float*    gaussb = carve((size_t)EN * GK);
  float*    aggb   = carve((size_t)NN * 64);
  float*    cntb   = carve(NN);
  float*    rtermb = carve((size_t)NN * 64);
  float*    h1     = carve((size_t)NN * 32);
  float*    h2b    = carve((size_t)NN * 64);
  unsigned* umax   = (unsigned*)carve((size_t)N2C * 64);
  float*    hp     = carve((size_t)N2C * 64);
  float*    psum   = carve((size_t)N2C * 2);
  float*    pcnt   = carve(N2C);
  float*    pos2   = carve((size_t)N2C * 2);
  unsigned* mxabs  = (unsigned*)carve(64);
  float*    ea2    = carve((size_t)E2C * 2);
  float*    h3     = carve((size_t)N2C * 64);
  float*    gsum   = carve((size_t)BG * 64);
  float*    gcnt   = carve(BG);
  // packed f16 weight fragments: (NC/16) tiles * NKS * 512 f16
  _Float16* g1p  = carveh(50 * 1 * 512);
  _Float16* g2p  = carveh(100 * 1 * 512);
  _Float16* g3p  = carveh(100 * 2 * 512);
  _Float16* rt1p = carveh(2 * 1 * 512);
  _Float16* rt2p = carveh(4 * 1 * 512);
  _Float16* rt3p = carveh(4 * 2 * 512);

  const int TB = 256;
  auto cdiv = [](int a, int b) { return (a + b - 1) / b; };

  // ---- pack all weight fragments ----
  pack_b_frags<<<cdiv(50 * 512, TB), TB, 0, stream>>>(g1, g1p, 16, 800, 1, 50 * 512);
  pack_b_frags<<<cdiv(100 * 512, TB), TB, 0, stream>>>(g2, g2p, 32, 1600, 1, 100 * 512);
  pack_b_frags<<<cdiv(100 * 2 * 512, TB), TB, 0, stream>>>(g3, g3p, 64, 1600, 2, 100 * 2 * 512);
  pack_b_frags<<<cdiv(2 * 512, TB), TB, 0, stream>>>(rt1, rt1p, 16, 32, 1, 2 * 512);
  pack_b_frags<<<cdiv(4 * 512, TB), TB, 0, stream>>>(rt2, rt2p, 32, 64, 1, 4 * 512);
  pack_b_frags<<<cdiv(4 * 2 * 512, TB), TB, 0, stream>>>(rt3, rt3p, 64, 64, 2, 4 * 2 * 512);

  // ---- layer 1: GMMConv 16 -> 32 ----
  gauss_kernel<<<cdiv(EN * GK, TB), TB, 0, stream>>>(eattr, mu1, sg1, gaussb, EN * GK);
  hipMemsetAsync(aggb, 0, (size_t)NN * 32 * sizeof(float), stream);
  hipMemsetAsync(cntb, 0, (size_t)NN * sizeof(float), stream);
  gmm_edge_wmma<16, 32><<<EN / 128, 256, 0, stream>>>(x, ei, gaussb, g1p, aggb, cntb, EN);
  gemm_root_wmma<16, 32><<<dim3(2, NN / 16), 32, 0, stream>>>(x, rt1p, rtermb);
  finalize_elu<32><<<cdiv(NN * 32, TB), TB, 0, stream>>>(aggb, cntb, rtermb, bb1, h1, NN * 32);

  // ---- layer 2: GMMConv 32 -> 64 ----
  gauss_kernel<<<cdiv(EN * GK, TB), TB, 0, stream>>>(eattr, mu2, sg2, gaussb, EN * GK);
  hipMemsetAsync(aggb, 0, (size_t)NN * 64 * sizeof(float), stream);
  hipMemsetAsync(cntb, 0, (size_t)NN * sizeof(float), stream);
  gmm_edge_wmma<32, 64><<<EN / 128, 256, 0, stream>>>(h1, ei, gaussb, g2p, aggb, cntb, EN);
  gemm_root_wmma<32, 64><<<dim3(4, NN / 16), 32, 0, stream>>>(h1, rt2p, rtermb);
  finalize_elu<64><<<cdiv(NN * 64, TB), TB, 0, stream>>>(aggb, cntb, rtermb, bb2, h2b, NN * 64);

  // ---- max_pool(cluster) + Cartesian ----
  hipMemsetAsync(umax, 0, (size_t)N2C * 64 * sizeof(unsigned), stream);
  hipMemsetAsync(psum, 0, (size_t)N2C * 2 * sizeof(float), stream);
  hipMemsetAsync(pcnt, 0, (size_t)N2C * sizeof(float), stream);
  hipMemsetAsync(mxabs, 0, sizeof(unsigned), stream);
  pool_max_scatter<<<cdiv(NN * 64, TB), TB, 0, stream>>>(h2b, clus, umax, NN * 64);
  pool_max_decode<<<cdiv(N2C * 64, TB), TB, 0, stream>>>(umax, hp, N2C * 64);
  pos_scatter<<<cdiv(NN, TB), TB, 0, stream>>>(pos, clus, psum, pcnt, NN);
  pos_mean<<<cdiv(N2C, TB), TB, 0, stream>>>(psum, pcnt, pos2, N2C);
  cart_max<<<cdiv(E2C, TB), TB, 0, stream>>>(pos2, ei2, mxabs, E2C);
  cart_write<<<cdiv(E2C, TB), TB, 0, stream>>>(pos2, ei2, mxabs, ea2, E2C);

  // ---- layer 3: GMMConv 64 -> 64 on pooled graph ----
  gauss_kernel<<<cdiv(E2C * GK, TB), TB, 0, stream>>>(ea2, mu3, sg3, gaussb, E2C * GK);
  hipMemsetAsync(aggb, 0, (size_t)N2C * 64 * sizeof(float), stream);
  hipMemsetAsync(cntb, 0, (size_t)N2C * sizeof(float), stream);
  gmm_edge_wmma<64, 64><<<E2C / 128, 256, 0, stream>>>(hp, ei2, gaussb, g3p, aggb, cntb, E2C);
  gemm_root_wmma<64, 64><<<dim3(4, N2C / 16), 32, 0, stream>>>(hp, rt3p, rtermb);
  finalize_elu<64><<<cdiv(N2C * 64, TB), TB, 0, stream>>>(aggb, cntb, rtermb, bb3, h3, N2C * 64);

  // ---- global mean pool + MLP head ----
  hipMemsetAsync(gsum, 0, (size_t)BG * 64 * sizeof(float), stream);
  hipMemsetAsync(gcnt, 0, (size_t)BG * sizeof(float), stream);
  graph_scatter<<<cdiv(N2C * 64, TB), TB, 0, stream>>>(h3, batch2, gsum, gcnt, N2C * 64);
  head_kernel<<<BG, 128, 0, stream>>>(gsum, gcnt, w1, b1v, w2, b2v, out);
}
